// SelfSupervised_6975026888742
// MI455X (gfx1250) — matrix operand order repaired and checked by reading, not compiled, not run on hardware
//
#include <hip/hip_runtime.h>

typedef __attribute__((ext_vector_type(16))) _Float16 v16h;
typedef __attribute__((ext_vector_type(8)))  float    v8f;
typedef __attribute__((ext_vector_type(4)))  int      v4i;

#define EPSV 1e-5f
#define Hh   128
#define Bb   1024
#define Tt   128
#define INN  75
#define KP0  96            // layer-0 K padded 75 -> 96 (multiple of 32)
#define OUTT 256
#define G4H  512           // 4*H
#define LPAD 40            // LDS row stride in halves: 80B = 20 banks, 16B aligned

// ---- CDNA5 async load-to-LDS (guarded; manual copy fallback) --------------
#if defined(__has_builtin)
#  if __has_builtin(__builtin_amdgcn_global_load_async_to_lds_b128)
#    define ASYNC_COPY 1
#  endif
#endif
#ifndef ASYNC_COPY
#  define ASYNC_COPY 0
#endif

#if ASYNC_COPY
__device__ __forceinline__ void async_cp16(const _Float16* g, _Float16* l)
{
    // param0: int4 addrspace(1)* (global src), param1: LDS dst, imm offset, cpol
    __builtin_amdgcn_global_load_async_to_lds_b128(
        (__attribute__((address_space(1))) v4i*)g,
        (__attribute__((address_space(3))) v4i*)l,
        0, 0);
}
__device__ __forceinline__ void async_wait_all()
{
#  if __has_builtin(__builtin_amdgcn_s_wait_asynccnt)
    __builtin_amdgcn_s_wait_asynccnt(0);
#  else
    asm volatile("s_wait_asynccnt 0x0" ::: "memory");
#  endif
}
#endif

// ---------------------------------------------------------------------------
// GEMM: C[M x N] = A[M x K] (f16, row-major, lda) @ Wt[N x K] (f16, row-major,
// ldwt = pre-transposed weights), f32 output + optional bias + optional f16
// copy. K must be a multiple of 32 (operands pre-padded). Block tile 64x64
// (grid.x = M/64, grid.y = N/64), 8 waves, each wave 16x32 via two 16x16
// V_WMMA_F32_16X16X32_F16. Double-buffered LDS; tiles staged as contiguous
// 16B chunks (async load-to-LDS when available).
// ---------------------------------------------------------------------------
__global__ __launch_bounds__(256)
void gemm_wmma_f16h(const _Float16* __restrict__ A, long lda,
                    const _Float16* __restrict__ Wt, long ldwt,
                    float* __restrict__ C, long ldc,
                    int Keff, const float* __restrict__ bias,
                    _Float16* __restrict__ CH)
{
    alignas(16) __shared__ _Float16 As[2][64 * LPAD];   // [m][k]
    alignas(16) __shared__ _Float16 Bs[2][64 * LPAD];   // [n][k]

    const int tid  = threadIdx.x;
    const int lane = tid & 31;
    const int wave = tid >> 5;
    const int wy   = wave & 3;
    const int wx   = wave >> 2;
    const int half = lane >> 4;
    const int l16  = lane & 15;
    const long rowbase = (long)blockIdx.x * 64;
    const long colbase = (long)blockIdx.y * 64;

    // staging: 256 threads x 16B chunk per tile; row = tid>>2, k-group = (tid&3)*8
    const int sr = tid >> 2;            // 0..63
    const int sc = (tid & 3) * 8;       // halves

    auto stage = [&](int kb, int buf) {
        const _Float16* ga = A  + (size_t)(rowbase + sr) * lda  + kb + sc;
        const _Float16* gb = Wt + (size_t)(colbase + sr) * ldwt + kb + sc;
        _Float16* la = &As[buf][sr * LPAD + sc];
        _Float16* lb = &Bs[buf][sr * LPAD + sc];
#if ASYNC_COPY
        async_cp16(ga, la);
        async_cp16(gb, lb);
#else
        #pragma unroll
        for (int u = 0; u < 8; ++u) { la[u] = ga[u]; lb[u] = gb[u]; }
#endif
    };

    v8f acc0 = {}, acc1 = {};
    stage(0, 0);
    int cur = 0;
    for (int kb = 0; kb < Keff; kb += 32) {
#if ASYNC_COPY
        async_wait_all();
#endif
        __syncthreads();
        if (kb + 32 < Keff) stage(kb + 32, cur ^ 1);   // overlap with WMMA

        // A fragment: lanes 0-15 -> K 0..15, lanes 16-31 -> K 16..31, lane%16 = M
        v16h af;
        #pragma unroll
        for (int v = 0; v < 8; ++v) {
            const int kk = half * 16 + 2 * v;
            const _Float16* p = &As[cur][(wy * 16 + l16) * LPAD + kk];
            af[2 * v]     = p[0];
            af[2 * v + 1] = p[1];
        }
        // B fragments: lanes 0-15 -> K 0..15, lanes 16-31 -> K 16..31, lane%16 = N
        v16h bf0, bf1;
        #pragma unroll
        for (int v = 0; v < 8; ++v) {
            const int kk = half * 16 + 2 * v;
            const _Float16* p0 = &Bs[cur][(wx * 32 + l16) * LPAD + kk];
            bf0[2 * v]     = p0[0];
            bf0[2 * v + 1] = p0[1];
            const _Float16* p1 = &Bs[cur][(wx * 32 + 16 + l16) * LPAD + kk];
            bf1[2 * v]     = p1[0];
            bf1[2 * v + 1] = p1[1];
        }
        acc0 = __builtin_amdgcn_wmma_f32_16x16x32_f16(false, af, false, bf0,
                                                      (short)0, acc0, false, false);
        acc1 = __builtin_amdgcn_wmma_f32_16x16x32_f16(false, af, false, bf1,
                                                      (short)0, acc1, false, false);
        cur ^= 1;
    }

    #pragma unroll
    for (int r = 0; r < 8; ++r) {
        const long grow = rowbase + wy * 16 + half * 8 + r;
        const long gc0  = colbase + wx * 32 + l16;
        const long gc1  = gc0 + 16;
        const float v0 = acc0[r] + (bias ? bias[gc0] : 0.0f);
        const float v1 = acc1[r] + (bias ? bias[gc1] : 0.0f);
        C[grow * ldc + gc0] = v0;
        C[grow * ldc + gc1] = v1;
        if (CH) {
            CH[grow * ldc + gc0] = (_Float16)v0;
            CH[grow * ldc + gc1] = (_Float16)v1;
        }
    }
}

__device__ __forceinline__ float sigm(float x) { return 1.0f / (1.0f + __expf(-x)); }

// ---------------------------------------------------------------------------
// Fused per-timestep BN-LSTM tail: block owns 4 j-columns (16 gate columns).
// Phase 1: BN stats of Yx/Yh for those columns (folded into affine ax,ah,cb).
// Phase 2: gates -> c1, o-gate; block-local c stats.
// Phase 3: h = sig(o)*tanh(BN(c1)) written as f16.
// grid.x = H/4 = 32, 256 threads. Deterministic tree reductions.
// ---------------------------------------------------------------------------
__global__ __launch_bounds__(256)
void fused_step(const float* __restrict__ Yx, long yxStride,
                const float* __restrict__ Yh,
                const float* __restrict__ gih, const float* __restrict__ bih,
                const float* __restrict__ ghh, const float* __restrict__ bhh,
                const float* __restrict__ bvec,
                const float* __restrict__ gcv, const float* __restrict__ bcv,
                float* __restrict__ c, _Float16* __restrict__ hOut)
{
    __shared__ float rsx[16][17], rqx[16][17], rsh[16][17], rqh[16][17];
    __shared__ float axL[16], ahL[16], cbL[16];
    __shared__ float csA[4][65], csB[4][65];
    __shared__ float mcL[4], rcL[4];

    const int j0 = blockIdx.x * 4;

    // ---- phase 1: per-gate-column BN stats over the batch ----
    {
        const int cx = threadIdx.x & 15;          // 16 gate cols of this block
        const int ry = threadIdx.x >> 4;          // 16 row lanes
        const int qi = cx >> 2, jj = cx & 3;
        const int col = qi * Hh + j0 + jj;
        float sx = 0, qx = 0, sh = 0, qh = 0;
        for (int r = ry; r < Bb; r += 16) {
            const float vx = Yx[(size_t)r * yxStride + col];
            const float vh = Yh[(size_t)r * G4H + col];
            sx += vx; qx += vx * vx; sh += vh; qh += vh * vh;
        }
        rsx[ry][cx] = sx; rqx[ry][cx] = qx; rsh[ry][cx] = sh; rqh[ry][cx] = qh;
        __syncthreads();
        for (int off = 8; off > 0; off >>= 1) {
            if (ry < off) {
                rsx[ry][cx] += rsx[ry + off][cx];
                rqx[ry][cx] += rqx[ry + off][cx];
                rsh[ry][cx] += rsh[ry + off][cx];
                rqh[ry][cx] += rqh[ry + off][cx];
            }
            __syncthreads();
        }
        if (ry == 0) {
            const float mxv = rsx[0][cx] / (float)Bb;
            const float rxv = rsqrtf(rqx[0][cx] / (float)Bb - mxv * mxv + EPSV);
            const float mhv = rsh[0][cx] / (float)Bb;
            const float rhv = rsqrtf(rqh[0][cx] / (float)Bb - mhv * mhv + EPSV);
            const float ax = gih[col] * rxv;
            const float ah = ghh[col] * rhv;
            axL[cx] = ax;
            ahL[cx] = ah;
            cbL[cx] = (bih[col] - ax * mxv) + (bhh[col] - ah * mhv) + bvec[col];
        }
        __syncthreads();
    }

    // ---- phase 2: cell update; each thread owns column jj=tid&3, 16 rows ----
    const int myjj = threadIdx.x & 3;
    const int myg  = threadIdx.x >> 2;            // 0..63 row group
    float c1r[16], ogr[16];
    float scc = 0, qcc = 0;
    #pragma unroll
    for (int e = 0; e < 16; ++e) {
        const int b = e * 64 + myg;
        float g[4];
        #pragma unroll
        for (int q = 0; q < 4; ++q) {
            const int gcol = q * Hh + j0 + myjj;
            const int lc   = q * 4 + myjj;
            g[q] = Yx[(size_t)b * yxStride + gcol] * axL[lc]
                 + Yh[(size_t)b * G4H + gcol] * ahL[lc] + cbL[lc];
        }
        const int cidx = b * Hh + j0 + myjj;
        const float c1 = sigm(g[0]) * c[cidx] + sigm(g[1]) * tanhf(g[3]);
        c[cidx] = c1;
        c1r[e] = c1;
        ogr[e] = g[2];
        scc += c1; qcc += c1 * c1;
    }
    csA[myjj][myg] = scc;
    csB[myjj][myg] = qcc;
    __syncthreads();
    for (int off = 32; off > 0; off >>= 1) {
        if (myg < off) {
            csA[myjj][myg] += csA[myjj][myg + off];
            csB[myjj][myg] += csB[myjj][myg + off];
        }
        __syncthreads();
    }
    if (threadIdx.x < 4) {
        const float m = csA[threadIdx.x][0] / (float)Bb;
        mcL[threadIdx.x] = m;
        rcL[threadIdx.x] = rsqrtf(csB[threadIdx.x][0] / (float)Bb - m * m + EPSV);
    }
    __syncthreads();

    // ---- phase 3: h output ----
    const int j = j0 + myjj;
    const float gcj = gcv[j], bcj = bcv[j];
    const float mcv = mcL[myjj], rcv = rcL[myjj];
    #pragma unroll
    for (int e = 0; e < 16; ++e) {
        const int b = e * 64 + myg;
        const float bn = gcj * (c1r[e] - mcv) * rcv + bcj;
        hOut[b * Hh + j] = (_Float16)(sigm(ogr[e]) * tanhf(bn));
    }
}

// ---- one-shot conversions -------------------------------------------------
// sequences (B,T,IN) f32 -> rows (b*T+t) x KP0 f16, zero-padded K tail
__global__ __launch_bounds__(256)
void seq_to_f16(const float* __restrict__ seq, _Float16* __restrict__ dst, int total)
{
    const int idx = blockIdx.x * 256 + threadIdx.x;
    if (idx >= total) return;
    const int r = idx / KP0, k = idx - r * KP0;
    const int b = r >> 7, t = r & (Tt - 1);
    dst[idx] = (k < INN) ? (_Float16)seq[(size_t)b * (Tt * INN) + t * INN + k]
                         : (_Float16)0.0f;
}

// W (K x N f32, row-major) -> Wt (N x Kp f16), zero-padded K tail
__global__ __launch_bounds__(256)
void w_to_f16t(const float* __restrict__ W, int K, int N, int Kp,
               _Float16* __restrict__ Wt, int total)
{
    const int idx = blockIdx.x * 256 + threadIdx.x;
    if (idx >= total) return;
    const int n = idx / Kp, k = idx - n * Kp;
    Wt[idx] = (k < K) ? (_Float16)W[(size_t)k * N + n] : (_Float16)0.0f;
}

// ---------------------------------------------------------------------------
extern "C" void kernel_launch(void* const* d_in, const int* in_sizes, int n_in,
                              void* d_out, int out_size, void* d_ws, size_t ws_size,
                              hipStream_t stream)
{
    (void)in_sizes; (void)n_in; (void)out_size;

    const float* seq = (const float*)d_in[0];
    const float* Wih[2]  = { (const float*)d_in[1],  (const float*)d_in[10] };
    const float* Whh[2]  = { (const float*)d_in[2],  (const float*)d_in[11] };
    const float* bv[2]   = { (const float*)d_in[3],  (const float*)d_in[12] };
    const float* gih[2]  = { (const float*)d_in[4],  (const float*)d_in[13] };
    const float* bih[2]  = { (const float*)d_in[5],  (const float*)d_in[14] };
    const float* ghh[2]  = { (const float*)d_in[6],  (const float*)d_in[15] };
    const float* bhh[2]  = { (const float*)d_in[7],  (const float*)d_in[16] };
    const float* gcv[2]  = { (const float*)d_in[8],  (const float*)d_in[17] };
    const float* bcv[2]  = { (const float*)d_in[9],  (const float*)d_in[18] };
    const float* fc_w  = (const float*)d_in[19];
    const float* fc_b  = (const float*)d_in[20];
    const float* dec_w = (const float*)d_in[21];
    const float* dec_b = (const float*)d_in[22];
    float* out = (float*)d_out;

    // ---- workspace carve-up (256B aligned blocks) ----
    char* base = (char*)d_ws;
    size_t off = 0;
    auto alloc = [&](size_t bytes) -> void* {
        void* p = base + off;
        off += (bytes + 255) & ~(size_t)255;
        return p;
    };
    _Float16* seqH   = (_Float16*)alloc((size_t)Bb * Tt * KP0 * 2);
    _Float16* WtIh0  = (_Float16*)alloc((size_t)G4H * KP0 * 2);
    _Float16* WtHh0  = (_Float16*)alloc((size_t)G4H * Hh * 2);
    _Float16* WtIh1  = (_Float16*)alloc((size_t)G4H * Hh * 2);
    _Float16* WtHh1  = (_Float16*)alloc((size_t)G4H * Hh * 2);
    _Float16* WtFc   = (_Float16*)alloc((size_t)OUTT * Hh * 2);
    _Float16* WtDec  = (_Float16*)alloc((size_t)(Tt * INN) * OUTT * 2);
    _Float16* hs0H   = (_Float16*)alloc((size_t)Tt * Bb * Hh * 2);
    _Float16* hbufH  = (_Float16*)alloc((size_t)Bb * Hh * 2);
    _Float16* zeroH  = (_Float16*)alloc((size_t)Bb * Hh * 2);
    _Float16* embH   = (_Float16*)alloc((size_t)Bb * OUTT * 2);
    float*    cbuf   = (float*)alloc((size_t)Bb * Hh * 4);
    float*    Yh     = (float*)alloc((size_t)Bb * G4H * 4);
    float*    Yx     = (float*)alloc((size_t)Bb * G4H * 4);
    float*    emb    = (float*)alloc((size_t)Bb * OUTT * 4);
    const size_t baseBytes  = off;
    const size_t hoistBytes = (size_t)Tt * Bb * G4H * 4;     // 256 MB
    const bool hoist = (baseBytes + hoistBytes) <= ws_size;
    float* Yxall = hoist ? (float*)alloc(hoistBytes) : (float*)0;

    const _Float16* WtIh[2] = { WtIh0, WtIh1 };
    const _Float16* WtHh[2] = { WtHh0, WtHh1 };

    const dim3 blk(256);

    // ---- one-shot operand conversion ----
    {
        const int nseq = Bb * Tt * KP0;
        seq_to_f16<<<(nseq + 255) / 256, blk, 0, stream>>>(seq, seqH, nseq);
        int n;
        n = G4H * KP0;        w_to_f16t<<<(n + 255) / 256, blk, 0, stream>>>(Wih[0], INN,  G4H,      KP0,  WtIh0, n);
        n = G4H * Hh;         w_to_f16t<<<(n + 255) / 256, blk, 0, stream>>>(Whh[0], Hh,   G4H,      Hh,   WtHh0, n);
        n = G4H * Hh;         w_to_f16t<<<(n + 255) / 256, blk, 0, stream>>>(Wih[1], Hh,   G4H,      Hh,   WtIh1, n);
        n = G4H * Hh;         w_to_f16t<<<(n + 255) / 256, blk, 0, stream>>>(Whh[1], Hh,   G4H,      Hh,   WtHh1, n);
        n = OUTT * Hh;        w_to_f16t<<<(n + 255) / 256, blk, 0, stream>>>(fc_w,   Hh,   OUTT,     Hh,   WtFc,  n);
        n = (Tt * INN) * OUTT; w_to_f16t<<<(n + 255) / 256, blk, 0, stream>>>(dec_w, OUTT, Tt * INN, OUTT, WtDec, n);
    }
    (void)hipMemsetAsync(zeroH, 0, (size_t)Bb * Hh * 2, stream);

    const dim3 stepGemmGrid(Bb / 64, G4H / 64);              // (16, 8)

    for (int layer = 0; layer < 2; ++layer) {
        const int Kin   = layer ? Hh : KP0;
        const long ldaX = layer ? Hh : KP0;
        const _Float16* Xin = layer ? hs0H : seqH;

        (void)hipMemsetAsync(cbuf, 0, (size_t)Bb * Hh * 4, stream);
        if (layer == 1)
            (void)hipMemsetAsync(hbufH, 0, (size_t)Bb * Hh * 2, stream);

        if (hoist) {
            // all timesteps' x @ Wih in one full-throughput GEMM
            gemm_wmma_f16h<<<dim3((Tt * Bb) / 64, G4H / 64), blk, 0, stream>>>(
                Xin, ldaX, WtIh[layer], ldaX, Yxall, G4H, Kin,
                (const float*)0, (_Float16*)0);
        }

        for (int t = 0; t < Tt; ++t) {
            const float* yxBase;
            long yxStride;
            if (hoist) {
                if (layer == 0) { yxBase = Yxall + (size_t)t * G4H;       yxStride = (long)Tt * G4H; }
                else            { yxBase = Yxall + (size_t)t * Bb * G4H;  yxStride = G4H; }
            } else {
                const _Float16* Ax = layer ? (hs0H + (size_t)t * Bb * Hh)
                                           : (seqH + (size_t)t * KP0);
                const long lda = layer ? Hh : (long)Tt * KP0;
                gemm_wmma_f16h<<<stepGemmGrid, blk, 0, stream>>>(
                    Ax, lda, WtIh[layer], ldaX, Yx, G4H, Kin,
                    (const float*)0, (_Float16*)0);
                yxBase = Yx; yxStride = G4H;
            }

            const _Float16* hprev =
                layer ? (const _Float16*)hbufH
                      : (t == 0 ? (const _Float16*)zeroH
                                : (const _Float16*)(hs0H + (size_t)(t - 1) * Bb * Hh));
            gemm_wmma_f16h<<<stepGemmGrid, blk, 0, stream>>>(
                hprev, Hh, WtHh[layer], Hh, Yh, G4H, Hh,
                (const float*)0, (_Float16*)0);

            _Float16* hOut = layer ? hbufH : (hs0H + (size_t)t * Bb * Hh);
            fused_step<<<Hh / 4, blk, 0, stream>>>(
                yxBase, yxStride, Yh,
                gih[layer], bih[layer], ghh[layer], bhh[layer], bv[layer],
                gcv[layer], bcv[layer], cbuf, hOut);
        }
    }

    // fc: emb = h_last @ fc_w + fc_b  (f32 + f16 copy for the next GEMM)
    gemm_wmma_f16h<<<dim3(Bb / 64, OUTT / 64), blk, 0, stream>>>(
        hbufH, Hh, WtFc, Hh, emb, OUTT, Hh, fc_b, embH);

    // decoder: out = emb @ dec_w + dec_b
    gemm_wmma_f16h<<<dim3(Bb / 64, (Tt * INN) / 64), blk, 0, stream>>>(
        embH, OUTT, WtDec, OUTT, out, Tt * INN, OUTT, dec_b, (_Float16*)0);
}
